// MSMLMultiHeadAttention_86500641342113
// MI455X (gfx1250) — matrix operand order repaired and checked by reading, not compiled, not run on hardware
//
#include <hip/hip_runtime.h>
#include <hip/hip_bf16.h>

// ---------------------------------------------------------------------------
// MSML linear attention, stage-1 (WIN=16), B=2 N=16384 C=256 W=1024
//   phiKw[b,m,c,w] = elu(X@Wk+bk)+2  (w-major relayout, bf16)
//   phiQ[b*1024+r, c] = elu(X@Wq+bq)+2  (first 1024 rows/batch, bf16)
//   KVt[b,m,d,c] = sum_w phiK[b,16w+m,c] * X[b,16w+m,d]   (bf16, transposed)
//   out[b,256w+16n+m,d] = sum_c phiQ[b,16w+n,c]*KVt[b,m,d,c] + p[b,w,n,m]
// Contractions: v_wmma_f32_16x16x32_bf16. bf16 tiles move via
// global_load_async_to_lds_b128 (ASYNCcnt) into double-buffered LDS so the
// DMA of tile k+1 overlaps the WMMAs of tile k; f32 tiles use a register
// pipeline (load k+1 early, convert/store after compute k).
// ---------------------------------------------------------------------------

typedef __bf16 bf16;
typedef __attribute__((ext_vector_type(16))) __bf16 v16bf;
typedef __attribute__((ext_vector_type(8)))  __bf16 v8bf;
typedef __attribute__((ext_vector_type(8)))  float  v8f;

#define LDS_PITCH 40   // halves per LDS row (32 data + 8 pad; 16B align, conflict-free)
#define A_TILE (128 * LDS_PITCH)
#define B_TILE (64 * LDS_PITCH)

__device__ __forceinline__ v16bf combine16(v8bf lo, v8bf hi) {
  v16bf r;
#pragma unroll
  for (int i = 0; i < 8; ++i) { r[i] = lo[i]; r[i + 8] = hi[i]; }
  return r;
}

// A fragment (16x32 bf16, M x K). lanes 0-15: row M=lane, K 0..7 / 16..23;
// lanes 16-31: row M=lane-16, K 8..15 / 24..31.
__device__ __forceinline__ v16bf load_a_frag(const bf16* lds, int row, int lane) {
  const int hb = lane >> 4;
  const bf16* base = lds + row * LDS_PITCH;
  v8bf lo = *(const v8bf*)(base + hb * 8);
  v8bf hi = *(const v8bf*)(base + 16 + hb * 8);
  return combine16(lo, hi);
}

// B fragment (32x16 bf16, K x N), LDS holds B column-major: lds[col][k].
__device__ __forceinline__ v16bf load_b_frag(const bf16* lds, int col, int lane) {
  const int hb = lane >> 4;
  const bf16* base = lds + col * LDS_PITCH;
  v8bf lo = *(const v8bf*)(base + hb * 8);
  v8bf hi = *(const v8bf*)(base + 16 + hb * 8);
  return combine16(lo, hi);
}

__device__ __forceinline__ v8f wmma_bf16(v16bf a, v16bf b, v8f c) {
  return __builtin_amdgcn_wmma_f32_16x16x32_bf16(
      false, a, false, b, (short)0, c, false, false);
}

__device__ __forceinline__ v8bf cvt8(float4 f0, float4 f1) {
  v8bf h;
  h[0] = (bf16)f0.x; h[1] = (bf16)f0.y; h[2] = (bf16)f0.z; h[3] = (bf16)f0.w;
  h[4] = (bf16)f1.x; h[5] = (bf16)f1.y; h[6] = (bf16)f1.z; h[7] = (bf16)f1.w;
  return h;
}

// --- CDNA5 async global->LDS copy (16B per lane), tracked by ASYNCcnt -------
__device__ __forceinline__ void async_ld_b128(const bf16* g, bf16* l) {
  const unsigned ldsOff = (unsigned)(unsigned long long)l;  // low 32b of LDS generic ptr = offset
  asm volatile("global_load_async_to_lds_b128 %0, %1, off"
               :: "v"(ldsOff), "v"(g) : "memory");
}
#define ASYNC_WAIT0() asm volatile("s_wait_asynccnt 0x0" ::: "memory")

// ---------------------------------------------------------------------------
// Kernel 1: projection + fused double-ELU feature map (register-pipelined).
//   y = X[grow,:]@W[:,col]+bias[col]; phi = elu(y)+2 -> bf16
// qmode=0: 32768 rows, store w-major phiKw[((b*16+m)*256+col)*1024+w]
// qmode=1: 2048 rows (grow remapped), store row-major phiQ[row*256+col]
// grid (rows/128, 4), block 256.
// ---------------------------------------------------------------------------
__global__ __launch_bounds__(256) void msml_proj_phi(
    const float* __restrict__ X, const float* __restrict__ W,
    const float* __restrict__ bias, bf16* __restrict__ outPhi, int qmode) {
  __shared__ bf16 sA[2][A_TILE];   // [row][k]
  __shared__ bf16 sB[2][B_TILE];   // [col][k]  (W^T tile)
  const int tid = threadIdx.x, lane = tid & 31, wave = tid >> 5;
  const int rowBlk = blockIdx.x * 128;
  const int colBase = blockIdx.y * 64;

  v8f acc[4] = {};

  const int ar  = tid >> 1;            // 0..127 A row
  const int akk = (tid & 1) * 16;      // 0 or 16
  const int lrow = rowBlk + ar;
  const int grow = qmode ? ((lrow >> 10) * 16384 + (lrow & 1023)) : lrow;
  const int bkr = tid >> 3;            // 0..31 W k-row
  const int bn8 = (tid & 7) * 8;       // 0..56 W col chunk

  const float* Xrow = X + (size_t)grow * 256 + akk;
  const float* Wrow = W + (size_t)bkr * 256 + colBase + bn8;

  float4 xr[4], wr[2];
  auto gload = [&](int k0) {
    const float4* sx = (const float4*)(Xrow + k0);
    xr[0] = sx[0]; xr[1] = sx[1]; xr[2] = sx[2]; xr[3] = sx[3];
    const float4* sw = (const float4*)(Wrow + (size_t)k0 * 256);
    wr[0] = sw[0]; wr[1] = sw[1];
  };
  auto lstore = [&](int buf) {
    *(v8bf*)(sA[buf] + ar * LDS_PITCH + akk)     = cvt8(xr[0], xr[1]);
    *(v8bf*)(sA[buf] + ar * LDS_PITCH + akk + 8) = cvt8(xr[2], xr[3]);
    float v[8] = {wr[0].x, wr[0].y, wr[0].z, wr[0].w, wr[1].x, wr[1].y, wr[1].z, wr[1].w};
#pragma unroll
    for (int i = 0; i < 8; ++i) sB[buf][(bn8 + i) * LDS_PITCH + bkr] = (bf16)v[i];
  };

  gload(0);
  lstore(0);
#pragma unroll
  for (int k = 0; k < 8; ++k) {
    if (k < 7) gload((k + 1) * 32);          // regs for tile k+1 in flight
    __syncthreads();                          // tile-k LDS visible; buf k+1 free
    const int cur = k & 1;
    v16bf a = load_a_frag(sA[cur], wave * 16 + (lane & 15), lane);
#pragma unroll
    for (int ct = 0; ct < 4; ++ct) {
      v16bf b = load_b_frag(sB[cur], ct * 16 + (lane & 15), lane);
      acc[ct] = wmma_bf16(a, b, acc[ct]);
    }
    if (k < 7) lstore(cur ^ 1);               // convert+store after compute
  }

  const int hb = lane >> 4, cl = lane & 15;
  const int tileRow0 = rowBlk + wave * 16;    // 16-aligned
#pragma unroll
  for (int ct = 0; ct < 4; ++ct) {
    const int coln = colBase + ct * 16 + cl;
    const float bv = bias[coln];
#pragma unroll
    for (int r = 0; r < 8; ++r) {
      const int row = tileRow0 + r + hb * 8;
      const float y = acc[ct][r] + bv;
      const float phi = (y > 0.f) ? (y + 2.f) : (__expf(y) + 1.f);  // elu(y)+2
      if (qmode) {
        outPhi[(size_t)row * 256 + coln] = (bf16)phi;
      } else {
        const int b = row >> 14, rl = row & 16383;
        const int w = rl >> 4, m = rl & 15;
        outPhi[(((size_t)b * 16 + m) * 256 + coln) * 1024 + w] = (bf16)phi;
      }
    }
  }
}

// ---------------------------------------------------------------------------
// Kernel 2: KVt[b,m,d,c] = sum_w phiKw[b,m,c,w] * X[b,16w+m,d]
// A tile async-DMA double-buffered; B tile register-pipelined f32->bf16.
// grid 256 (= 32 (b,m) * 2 cBlk * 4 dBlk), block 256.
// ---------------------------------------------------------------------------
__global__ __launch_bounds__(256) void msml_kv(
    const bf16* __restrict__ phiKw, const float* __restrict__ X,
    bf16* __restrict__ KVt) {
  __shared__ bf16 sA[2][A_TILE];   // [c_local][j]
  __shared__ bf16 sB[2][B_TILE];   // [d_local][j]
  const int tid = threadIdx.x, lane = tid & 31, wave = tid >> 5;
  const int bm = blockIdx.x >> 3;           // b*16+m
  const int sub = blockIdx.x & 7;
  const int c0 = (sub >> 2) * 128;
  const int d0 = (sub & 3) * 64;
  const int b = bm >> 4, m = bm & 15;
  const size_t rowbase = (size_t)b * 16384 + m;   // X row = rowbase + 16*(w0+j)

  v8f acc[4] = {};
  const int j   = tid >> 3;        // 0..31 (window index within K-step)
  const int d8  = (tid & 7) * 8;   // V column chunk
  const int arow0 = tid >> 2, ach = (tid & 3) * 8;     // A chunks (2/thread)
  const int arow1 = (tid + 256) >> 2;

  const bf16* aBase = phiKw + ((size_t)bm * 256 + c0) * 1024;
  const float* bBase = X + (rowbase + (size_t)j * 16) * 256 + d0 + d8;

  auto issueA = [&](int w0, int buf) {
    async_ld_b128(aBase + (size_t)arow0 * 1024 + w0 + ach, sA[buf] + arow0 * LDS_PITCH + ach);
    async_ld_b128(aBase + (size_t)arow1 * 1024 + w0 + ach, sA[buf] + arow1 * LDS_PITCH + ach);
  };
  float4 br[2];
  auto gloadB = [&](int w0) {
    const float4* s = (const float4*)(bBase + (size_t)w0 * 16 * 256);
    br[0] = s[0]; br[1] = s[1];
  };
  auto lstoreB = [&](int buf) {
    float v[8] = {br[0].x, br[0].y, br[0].z, br[0].w, br[1].x, br[1].y, br[1].z, br[1].w};
#pragma unroll
    for (int i = 0; i < 8; ++i) sB[buf][(d8 + i) * LDS_PITCH + j] = (bf16)v[i];
  };

  issueA(0, 0);
  gloadB(0);
  lstoreB(0);
  for (int k = 0; k < 32; ++k) {
    const int cur = k & 1;
    if (k < 31) gloadB((k + 1) * 32);      // f32 loads for tile k+1 in flight
    ASYNC_WAIT0();                          // my tile-k DMA done
    __syncthreads();                        // everyone's tile k ready; buf k+1 free
    if (k < 31) issueA((k + 1) * 32, cur ^ 1);  // DMA of k+1 overlaps WMMAs of k
    v16bf a = load_a_frag(sA[cur], wave * 16 + (lane & 15), lane);
#pragma unroll
    for (int ct = 0; ct < 4; ++ct) {
      v16bf bfrag = load_b_frag(sB[cur], ct * 16 + (lane & 15), lane);
      acc[ct] = wmma_bf16(a, bfrag, acc[ct]);
    }
    if (k < 31) lstoreB(cur ^ 1);
  }

  const int hb = lane >> 4, cl = lane & 15;
  const size_t base = (size_t)bm * 256 * 256;
#pragma unroll
  for (int ct = 0; ct < 4; ++ct) {
#pragma unroll
    for (int r = 0; r < 8; ++r) {
      const int crow = c0 + wave * 16 + r + hb * 8;
      const int dcol = d0 + ct * 16 + cl;
      KVt[base + (size_t)dcol * 256 + crow] = (bf16)acc[ct][r];   // transposed store
    }
  }
}

// ---------------------------------------------------------------------------
// Kernel 3: out[b,256w+16n+m,d] = sum_c phiQ[b*1024+16w+n,c]*KVt[b,m,d,c] + p
//           p = embed_qk[b,w,n,m,:] . Wp + bp
// Both tiles async-DMA, double-buffered. grid 1024, block 256.
// ---------------------------------------------------------------------------
__global__ __launch_bounds__(256) void msml_att(
    const bf16* __restrict__ phiQ, const bf16* __restrict__ KVt,
    const float* __restrict__ EQ, const float* __restrict__ Wp,
    const float* __restrict__ bp, float* __restrict__ Out) {
  __shared__ bf16 sA[2][A_TILE];   // [row][k] phiQ
  __shared__ bf16 sB[2][B_TILE];   // [d][k]   KVt rows
  const int tid = threadIdx.x, lane = tid & 31, wave = tid >> 5;
  const int bm = blockIdx.x >> 5;
  const int sub = blockIdx.x & 31;
  const int r0 = (sub >> 2) * 128;
  const int d0 = (sub & 3) * 64;
  const int b = bm >> 4, m = bm & 15;

  v8f acc[4] = {};
  const int arow0 = tid >> 2, ach = (tid & 3) * 8;     // A chunks (2/thread)
  const int arow1 = (tid + 256) >> 2;
  const int brow  = tid >> 2;                          // B chunks (1/thread)

  const bf16* aBase = phiQ + ((size_t)b * 1024 + r0) * 256;
  const bf16* bBase = KVt + ((size_t)bm * 256 + d0) * 256;

  auto issueT = [&](int k0, int buf) {
    async_ld_b128(aBase + (size_t)arow0 * 256 + k0 + ach, sA[buf] + arow0 * LDS_PITCH + ach);
    async_ld_b128(aBase + (size_t)arow1 * 256 + k0 + ach, sA[buf] + arow1 * LDS_PITCH + ach);
    async_ld_b128(bBase + (size_t)brow * 256 + k0 + ach, sB[buf] + brow * LDS_PITCH + ach);
  };

  // prefetch the positional-bias rows this block needs in the epilogue
  __builtin_prefetch((const void*)(EQ + ((((size_t)b * 1024 + (r0 >> 4)) * 16) * 16 + m) * 3), 0, 0);

  issueT(0, 0);
#pragma unroll
  for (int k = 0; k < 8; ++k) {
    const int cur = k & 1;
    ASYNC_WAIT0();                          // my tile-k DMA done
    __syncthreads();                        // everyone's tile k ready; buf k+1 free
    if (k < 7) issueT((k + 1) * 32, cur ^ 1);   // DMA of k+1 overlaps WMMAs of k
    v16bf a = load_a_frag(sA[cur], wave * 16 + (lane & 15), lane);
#pragma unroll
    for (int ct = 0; ct < 4; ++ct) {
      v16bf bfrag = load_b_frag(sB[cur], ct * 16 + (lane & 15), lane);
      acc[ct] = wmma_bf16(a, bfrag, acc[ct]);
    }
  }

  const float wp0 = Wp[0], wp1 = Wp[1], wp2 = Wp[2], bpv = bp[0];
  const int hb = lane >> 4, cl = lane & 15;
  float pv[8];
#pragma unroll
  for (int r = 0; r < 8; ++r) {
    const int rl = r0 + wave * 16 + r + hb * 8;  // 0..1023
    const int w = rl >> 4, nn = rl & 15;
    const float* e = EQ + ((((size_t)b * 1024 + w) * 16 + nn) * 16 + m) * 3;
    pv[r] = e[0] * wp0 + e[1] * wp1 + e[2] * wp2 + bpv;
  }
#pragma unroll
  for (int ct = 0; ct < 4; ++ct) {
#pragma unroll
    for (int r = 0; r < 8; ++r) {
      const int rl = r0 + wave * 16 + r + hb * 8;
      const int w = rl >> 4, nn = rl & 15;
      const int orow = (w << 8) + (nn << 4) + m;
      Out[((size_t)b * 16384 + orow) * 256 + d0 + ct * 16 + cl] = acc[ct][r] + pv[r];
    }
  }
}

// ---------------------------------------------------------------------------
extern "C" void kernel_launch(void* const* d_in, const int* in_sizes, int n_in,
                              void* d_out, int out_size, void* d_ws, size_t ws_size,
                              hipStream_t stream) {
  const float* X  = (const float*)d_in[0];   // (2,16384,256)
  const float* EQ = (const float*)d_in[1];   // (2,1024,16,16,3)
  const float* Wq = (const float*)d_in[2];   // (256,256)
  const float* bq = (const float*)d_in[3];   // (256)
  const float* Wk = (const float*)d_in[4];   // (256,256)
  const float* bk = (const float*)d_in[5];   // (256)
  const float* Wp = (const float*)d_in[6];   // (3,1)
  const float* bp = (const float*)d_in[7];   // (1)
  float* Out = (float*)d_out;                // (2,16384,256) f32

  bf16* phiKw = (bf16*)d_ws;                                  // [32][256][1024]
  bf16* phiQ  = phiKw + (size_t)32 * 256 * 1024;              // [2048][256]
  bf16* KVt   = phiQ + (size_t)2048 * 256;                    // [32][256][256]

  // 1) phiK (w-major) over all 32768 rows
  msml_proj_phi<<<dim3(256, 4), 256, 0, stream>>>(X, Wk, bk, phiKw, 0);
  // 2) phiQ over first 1024 rows of each batch (2048 rows, remapped)
  msml_proj_phi<<<dim3(16, 4), 256, 0, stream>>>(X, Wq, bq, phiQ, 1);
  // 3) KV accumulation: 32 GEMMs of 256x256, K=1024 (async A-tiles, pipelined)
  msml_kv<<<dim3(256), 256, 0, stream>>>(phiKw, X, KVt);
  // 4) attention output GEMM + positional bias (fully async, pipelined)
  msml_att<<<dim3(1024), 256, 0, stream>>>(phiQ, KVt, EQ, Wp, bp, Out);
}
